// Attention_43155831390219
// MI455X (gfx1250) — matrix operand order repaired and checked by reading, not compiled, not run on hardware
//
#include <hip/hip_runtime.h>
#include <hip/hip_fp16.h>

// ---------------------------------------------------------------------------
// MI455X (gfx1250), wave32, v_wmma_f32_16x16x32_f16.
// B=16, H=W=56, N=3136, C=256, h2=4 heads/branch, hd=32, SR=4 (Nk=196),
// WS=7 (64 windows of 49).
// All WMMA fragments are contiguous 128-bit accesses; the GEMM is LDS-free
// and software-pipelined (register double-buffering) so WMMAs overlap the
// next K-step's global loads.  Softmax uses v_rcp_f32 instead of divides.
// ---------------------------------------------------------------------------

typedef _Float16 f16;
typedef __attribute__((ext_vector_type(16))) _Float16 v16h;
typedef __attribute__((ext_vector_type(8)))  _Float16 v8h;
typedef __attribute__((ext_vector_type(8)))  float    v8f;

#define V8FZ {0.f,0.f,0.f,0.f,0.f,0.f,0.f,0.f}

__device__ __forceinline__ v8f wmma32(v16h a, v16h b, v8f c) {
  return __builtin_amdgcn_wmma_f32_16x16x32_f16(false, a, false, b, (short)0, c, false, false);
}
__device__ __forceinline__ v16h cat8(v8h lo, v8h hi) {
  return __builtin_shufflevector(lo, hi, 0,1,2,3,4,5,6,7,8,9,10,11,12,13,14,15);
}

// ---------------------------------------------------------------------------
// GEMM: C[M,N] = A(f16 row-major) @ Bp(frag-packed f16) + bias.
// Block = 128 thr (4 waves), tile 64x64, K-step 32.  No LDS.
// Software-pipelined: loads for ks+1 issued before WMMAs for ks.
// Output: f32 (Cf) or f16 row-major (Ch) -- exactly one non-null.
// ---------------------------------------------------------------------------
__global__ __launch_bounds__(128) void gemm_rm(
    const f16* __restrict__ A, const f16* __restrict__ Bp,
    const float* __restrict__ bias,
    float* __restrict__ Cf, f16* __restrict__ Ch,
    int M, int N, int K) {
  const int tid = threadIdx.x, wave = tid >> 5, lane = tid & 31;
  const int g = lane >> 4, cl = lane & 15;
  const int m0 = blockIdx.y * 64, n0 = blockIdx.x * 64;
  const int kSteps = K >> 5, nTiles = N >> 4;
  const int arow = m0 + wave * 16 + cl;               // A-frag: row per lane
  const f16* aBase = A + (size_t)arow * K + g * 8;
  const f16* bBase = Bp + (size_t)(n0 >> 4) * 512 + lane * 16;
  v8f acc[4] = {V8FZ, V8FZ, V8FZ, V8FZ};

  v16h a = cat8(*(const v8h*)(aBase), *(const v8h*)(aBase + 16));
  v16h b0 = *(const v16h*)(bBase + 0 * 512);
  v16h b1 = *(const v16h*)(bBase + 1 * 512);
  v16h b2 = *(const v16h*)(bBase + 2 * 512);
  v16h b3 = *(const v16h*)(bBase + 3 * 512);

#pragma unroll 2
  for (int ks = 0; ks < kSteps - 1; ++ks) {
    if (ks + 2 < kSteps)
      __builtin_prefetch(aBase + (size_t)(ks + 2) * 32, 0, 3);
    const f16* an = aBase + (size_t)(ks + 1) * 32;
    const f16* bk = bBase + (size_t)(ks + 1) * nTiles * 512;
    v16h na = cat8(*(const v8h*)an, *(const v8h*)(an + 16));
    v16h c0 = *(const v16h*)(bk + 0 * 512);
    v16h c1 = *(const v16h*)(bk + 1 * 512);
    v16h c2 = *(const v16h*)(bk + 2 * 512);
    v16h c3 = *(const v16h*)(bk + 3 * 512);
    acc[0] = wmma32(a, b0, acc[0]);
    acc[1] = wmma32(a, b1, acc[1]);
    acc[2] = wmma32(a, b2, acc[2]);
    acc[3] = wmma32(a, b3, acc[3]);
    a = na; b0 = c0; b1 = c1; b2 = c2; b3 = c3;
  }
  acc[0] = wmma32(a, b0, acc[0]);
  acc[1] = wmma32(a, b1, acc[1]);
  acc[2] = wmma32(a, b2, acc[2]);
  acc[3] = wmma32(a, b3, acc[3]);

#pragma unroll
  for (int nt = 0; nt < 4; ++nt) {
    int col = n0 + nt * 16 + cl;
    float bv = bias ? bias[col] : 0.f;
#pragma unroll
    for (int r = 0; r < 8; ++r) {
      int row = m0 + wave * 16 + r + 8 * g;
      float v = acc[nt][r] + bv;
      if (Ch) Ch[(size_t)row * N + col] = (f16)v;
      else    Cf[(size_t)row * N + col] = v;
    }
  }
}

// ---------------------------------------------------------------------------
// Pack a K x N f32 weight into fragment layout [K/32][N/16][lane(32)][16] f16.
// ---------------------------------------------------------------------------
__global__ __launch_bounds__(256) void pack_wB(const float* __restrict__ W,
                                               f16* __restrict__ dst,
                                               int Kst, int Ntl, int N) {
  int idx = blockIdx.x * 256 + threadIdx.x;
  int total = Kst * Ntl * 64;
  if (idx >= total) return;
  int half = idx & 1, lane = (idx >> 1) & 31, t2 = idx >> 6;
  int nt = t2 % Ntl, ks = t2 / Ntl;
  int g = lane >> 4, cl = lane & 15;
  int kb = ks * 32 + g * 8 + half * 16;
  int n = nt * 16 + cl;
  v8h o;
#pragma unroll
  for (int i = 0; i < 8; ++i) o[i] = (f16)W[(size_t)(kb + i) * N + n];
  *(v8h*)(dst + ((size_t)(ks * Ntl + nt)) * 512 + lane * 16 + half * 8) = o;
}

// sr_w (co,ci,kh,kw) -> frag-packed [128][16][512] with k=(kh*4+kw)*256+ci, n=co
__global__ __launch_bounds__(256) void pack_srwB(const float* __restrict__ srw,
                                                 f16* __restrict__ dst) {
  int idx = blockIdx.x * 256 + threadIdx.x;          // 128*16*64 = 131072
  if (idx >= 131072) return;
  int half = idx & 1, lane = (idx >> 1) & 31, t2 = idx >> 6;
  int nt = t2 & 15, ks = t2 >> 4;
  int g = lane >> 4, cl = lane & 15;
  int kb = ks * 32 + g * 8 + half * 16;              // 8-aligned, same khkw
  int co = nt * 16 + cl;
  int khkw = kb >> 8, ci0 = kb & 255;
  int kh = khkw >> 2, kw = khkw & 3;
  v8h o;
#pragma unroll
  for (int i = 0; i < 8; ++i)
    o[i] = (f16)srw[((size_t)(co * 256 + ci0 + i) * 4 + kh) * 4 + kw];
  *(v8h*)(dst + ((size_t)(ks * 16 + nt)) * 512 + lane * 16 + half * 8) = o;
}

// f32 -> f16 convert, 8 elements per thread (b128 in, b128 out)
__global__ __launch_bounds__(256) void cvt8(const float* __restrict__ s,
                                            f16* __restrict__ d, int n8) {
  int i = blockIdx.x * 256 + threadIdx.x;
  if (i >= n8) return;
  const float* p = s + (size_t)i * 8;
  v8h o;
#pragma unroll
  for (int j = 0; j < 8; ++j) o[j] = (f16)p[j];
  *(v8h*)(d + (size_t)i * 8) = o;
}

// (a+b) -> f16, 8 per thread (proj GEMM input: y + lepe)
__global__ __launch_bounds__(256) void addcvt8(const float* __restrict__ a,
                                               const float* __restrict__ b,
                                               f16* __restrict__ d, int n8) {
  int i = blockIdx.x * 256 + threadIdx.x;
  if (i >= n8) return;
  const float* pa = a + (size_t)i * 8;
  const float* pb = b + (size_t)i * 8;
  v8h o;
#pragma unroll
  for (int j = 0; j < 8; ++j) o[j] = (f16)(pa[j] + pb[j]);
  *(v8h*)(d + (size_t)i * 8) = o;
}

// SR conv im2col: srA16[p = b*196 + ph*14+pw][k=(kh*4+kw)*256+ci] (f16 row-major)
__global__ __launch_bounds__(256) void gather_srA(const float* __restrict__ x,
                                                  f16* __restrict__ dst) {
  int idx = blockIdx.x * 256 + threadIdx.x;          // 3136*512 = 1605632
  if (idx >= 3136 * 512) return;
  int ch = idx & 511, p = idx >> 9;
  int k0 = ch * 8;
  int b = p / 196, pp = p % 196;
  int ph = pp / 14, pw = pp % 14;
  int khkw = k0 >> 8, ci0 = k0 & 255;
  int kh = khkw >> 2, kw = khkw & 3;
  int pix = (4 * ph + kh) * 56 + 4 * pw + kw;
  const float* s = x + ((size_t)b * 3136 + pix) * 256 + ci0;
  v8h o;
#pragma unroll
  for (int i = 0; i < 8; ++i) o[i] = (f16)s[i];
  *(v8h*)(dst + (size_t)p * 4096 + k0) = o;
}

// Vt1[b][h][d(32)][key(224)] f16, zero-padded keys >=196  (global-branch V^T)
__global__ __launch_bounds__(256) void pack_Vt1(const f16* __restrict__ kv1h,
                                                f16* __restrict__ Vt1) {
  int idx = blockIdx.x * 256 + threadIdx.x;          // 16*4*32*28 = 57344
  if (idx >= 57344) return;
  int ch = idx % 28, d = (idx / 28) & 31, h = (idx / (28 * 32)) & 3,
      b = idx / (28 * 32 * 4);
  v8h o;
#pragma unroll
  for (int i = 0; i < 8; ++i) {
    int key = ch * 8 + i;
    o[i] = (key < 196)
               ? kv1h[((size_t)b * 196 + key) * 256 + 128 + h * 32 + d]
               : (f16)0.f;
  }
  *(v8h*)(Vt1 + (((size_t)(b * 4 + h) * 32 + d) * 224) + ch * 8) = o;
}

// Vt2[(b*4+h)*64+win][d(32)][key(64)] f16, window-gathered, zero key>=49
__global__ __launch_bounds__(256) void pack_Vt2(const f16* __restrict__ kv2h,
                                                f16* __restrict__ Vt2) {
  int idx = blockIdx.x * 256 + threadIdx.x;          // 2^20
  if (idx >= (1 << 20)) return;
  int ch = idx & 7, d = (idx >> 3) & 31, win = (idx >> 8) & 63,
      h = (idx >> 14) & 3, b = idx >> 16;
  int wy = win >> 3, wx = win & 7;
  v8h o;
#pragma unroll
  for (int i = 0; i < 8; ++i) {
    int key = ch * 8 + i;
    f16 v = (f16)0.f;
    if (key < 49) {
      int ky = key / 7, kx = key % 7;
      int n = (wy * 7 + ky) * 56 + wx * 7 + kx;
      v = kv2h[((size_t)b * 3136 + n) * 256 + 128 + h * 32 + d];
    }
    o[i] = v;
  }
  *(v8h*)(Vt2 + ((((size_t)(b * 4 + h) * 64 + win) * 32 + d) * 64) + ch * 8) = o;
}

// Depthwise 3x3 on t (B,N,C) with zero pad.
__global__ __launch_bounds__(256) void lepe_conv(
    const float* __restrict__ t, const float* __restrict__ cw,
    const float* __restrict__ cb, float* __restrict__ lepe) {
  int bn = blockIdx.x, c = threadIdx.x;
  int b = bn / 3136, n = bn % 3136;
  int h = n / 56, w = n % 56;
  float acc = cb[c];
#pragma unroll
  for (int dh = -1; dh <= 1; ++dh) {
    int hh = h + dh;
    if (hh < 0 || hh >= 56) continue;
#pragma unroll
    for (int dw = -1; dw <= 1; ++dw) {
      int ww = w + dw;
      if (ww < 0 || ww >= 56) continue;
      acc += cw[c * 9 + (dh + 1) * 3 + (dw + 1)] *
             t[((size_t)b * 3136 + hh * 56 + ww) * 256 + c];
    }
  }
  lepe[(size_t)bn * 256 + c] = acc;
}

// LayerNorm(256) + exact GELU, f16 output (GEMM A operand).
__global__ __launch_bounds__(256) void ln_gelu(
    const float* __restrict__ xs, const float* __restrict__ gam,
    const float* __restrict__ bet, f16* __restrict__ out) {
  int row = blockIdx.x, tid = threadIdx.x;
  float v = xs[(size_t)row * 256 + tid];
  float s = v, s2 = v * v;
#pragma unroll
  for (int m = 16; m >= 1; m >>= 1) {
    s += __shfl_xor(s, m, 32);
    s2 += __shfl_xor(s2, m, 32);
  }
  __shared__ float r1[8], r2[8];
  int wv = tid >> 5, ln = tid & 31;
  if (ln == 0) { r1[wv] = s; r2[wv] = s2; }
  __syncthreads();
  float ts = 0.f, ts2 = 0.f;
#pragma unroll
  for (int i = 0; i < 8; ++i) { ts += r1[i]; ts2 += r2[i]; }
  float mu = ts * (1.f / 256.f);
  float var = ts2 * (1.f / 256.f) - mu * mu;
  float y = (v - mu) * rsqrtf(var + 1e-5f) * gam[tid] + bet[tid];
  out[(size_t)row * 256 + tid] =
      (f16)(y * 0.5f * (1.f + erff(y * 0.70710678118654752f)));
}

// ---------------------------------------------------------------------------
// Global attention: (64-q tile, head, batch).  Keys 196 -> 224 padded.
// K-frags straight from kv1h (v16h); V-frags from Vt1 (v16h); P via LDS.
// ---------------------------------------------------------------------------
__global__ __launch_bounds__(128) void attn_global(
    const f16* __restrict__ q1h, const f16* __restrict__ kv1h,
    const f16* __restrict__ Vt1, float* __restrict__ y,
    float* __restrict__ gm) {
  __shared__ __align__(32) f16 Ps[4][16][240];
  const int tid = threadIdx.x, wave = tid >> 5, lane = tid & 31;
  const int g = lane >> 4, cl = lane & 15;
  const int qbase = blockIdx.x * 64, h = blockIdx.y, b = blockIdx.z;

  const int qrow = b * 3136 + qbase + wave * 16 + cl;
  const f16* qp = q1h + (size_t)qrow * 128 + h * 32 + g * 16;
  v16h a = cat8(*(const v8h*)qp, *(const v8h*)(qp + 8));

  v8f S[14];
#pragma unroll
  for (int nt = 0; nt < 14; ++nt) {
    int key = nt * 16 + cl;
    int kr = key < 196 ? key : 195;                   // clamp, masked below
    const f16* kp = kv1h + ((size_t)b * 196 + kr) * 256 + h * 32 + g * 16;
    v16h bf = *(const v16h*)kp;
    v8f z = V8FZ;
    S[nt] = wmma32(a, bf, z);
  }

  const float scale = 0.17677669529663689f;           // 32^-0.5
  float rinv[8];
#pragma unroll
  for (int r = 0; r < 8; ++r) {
    float mx = -1e30f;
#pragma unroll
    for (int nt = 0; nt < 14; ++nt) {
      int key = nt * 16 + cl;
      float sv = (key < 196) ? S[nt][r] * scale : -1e30f;
      S[nt][r] = sv;
      mx = fmaxf(mx, sv);
    }
    mx = fmaxf(mx, __shfl_xor(mx, 1, 32));
    mx = fmaxf(mx, __shfl_xor(mx, 2, 32));
    mx = fmaxf(mx, __shfl_xor(mx, 4, 32));
    mx = fmaxf(mx, __shfl_xor(mx, 8, 32));
    float sum = 0.f;
#pragma unroll
    for (int nt = 0; nt < 14; ++nt) {
      float p = __expf(S[nt][r] - mx);
      S[nt][r] = p;
      sum += p;
    }
    sum += __shfl_xor(sum, 1, 32);
    sum += __shfl_xor(sum, 2, 32);
    sum += __shfl_xor(sum, 4, 32);
    sum += __shfl_xor(sum, 8, 32);
    rinv[r] = __builtin_amdgcn_rcpf(sum);             // v_rcp_f32, no div
  }

#pragma unroll
  for (int nt = 0; nt < 14; ++nt) {                   // gm += col sums of attn
    float cs = 0.f;
#pragma unroll
    for (int r = 0; r < 8; ++r) cs += S[nt][r] * rinv[r];
    cs += __shfl_xor(cs, 16, 32);
    int key = nt * 16 + cl;
    if (lane < 16 && key < 196) atomicAdd(&gm[b * 196 + key], cs);
  }

#pragma unroll
  for (int nt = 0; nt < 14; ++nt)                     // P transpose via LDS
#pragma unroll
    for (int r = 0; r < 8; ++r)
      Ps[wave][r + 8 * g][nt * 16 + cl] = (f16)S[nt][r];
  __syncthreads();

  v8f o[2] = {V8FZ, V8FZ};
#pragma unroll
  for (int kt = 0; kt < 7; ++kt) {
    const f16* pp = &Ps[wave][cl][kt * 32 + g * 16];
    v16h pa = cat8(*(const v8h*)pp, *(const v8h*)(pp + 8));
#pragma unroll
    for (int dt = 0; dt < 2; ++dt) {
      const f16* vp = Vt1 + (((size_t)(b * 4 + h) * 32 + dt * 16 + cl) * 224) +
                      kt * 32 + g * 16;
      o[dt] = wmma32(pa, *(const v16h*)vp, o[dt]);
    }
  }
#pragma unroll
  for (int dt = 0; dt < 2; ++dt) {
    int col = h * 32 + dt * 16 + cl;
#pragma unroll
    for (int r = 0; r < 8; ++r) {
      int row = qbase + wave * 16 + r + 8 * g;
      y[((size_t)b * 3136 + row) * 256 + col] = o[dt][r] * rinv[r];
    }
  }
}

// ---------------------------------------------------------------------------
// Local attention: (window, head, batch); 49x49 padded to 64.
// ---------------------------------------------------------------------------
__global__ __launch_bounds__(128) void attn_local(
    const f16* __restrict__ q2h, const f16* __restrict__ kv2h,
    const f16* __restrict__ Vt2, float* __restrict__ y,
    float* __restrict__ lm) {
  __shared__ __align__(32) f16 Ps[4][16][64];
  const int tid = threadIdx.x, wave = tid >> 5, lane = tid & 31;
  const int g = lane >> 4, cl = lane & 15;
  const int win = blockIdx.x, wy = win >> 3, wx = win & 7;
  const int h = blockIdx.y, b = blockIdx.z;

  v16h a;
  {
    int q = wave * 16 + cl;
    int qq = q < 49 ? q : 48;                         // rows >=49 discarded
    int qy = qq / 7, qx = qq % 7;
    int n = (wy * 7 + qy) * 56 + wx * 7 + qx;
    const f16* qp = q2h + ((size_t)b * 3136 + n) * 128 + h * 32 + g * 16;
    a = cat8(*(const v8h*)qp, *(const v8h*)(qp + 8));
  }

  v8f S[4];
#pragma unroll
  for (int nt = 0; nt < 4; ++nt) {
    int key = nt * 16 + cl;
    int kk = key < 49 ? key : 48;
    int ky = kk / 7, kx = kk % 7;
    int n = (wy * 7 + ky) * 56 + wx * 7 + kx;
    const f16* kp = kv2h + ((size_t)b * 3136 + n) * 256 + h * 32 + g * 16;
    v8f z = V8FZ;
    S[nt] = wmma32(a, *(const v16h*)kp, z);
  }

  const float scale = 0.17677669529663689f;
  float rinv[8];
#pragma unroll
  for (int r = 0; r < 8; ++r) {
    float mx = -1e30f;
#pragma unroll
    for (int nt = 0; nt < 4; ++nt) {
      int key = nt * 16 + cl;
      float sv = (key < 49) ? S[nt][r] * scale : -1e30f;
      S[nt][r] = sv;
      mx = fmaxf(mx, sv);
    }
    mx = fmaxf(mx, __shfl_xor(mx, 1, 32));
    mx = fmaxf(mx, __shfl_xor(mx, 2, 32));
    mx = fmaxf(mx, __shfl_xor(mx, 4, 32));
    mx = fmaxf(mx, __shfl_xor(mx, 8, 32));
    float sum = 0.f;
#pragma unroll
    for (int nt = 0; nt < 4; ++nt) {
      float p = __expf(S[nt][r] - mx);
      S[nt][r] = p;
      sum += p;
    }
    sum += __shfl_xor(sum, 1, 32);
    sum += __shfl_xor(sum, 2, 32);
    sum += __shfl_xor(sum, 4, 32);
    sum += __shfl_xor(sum, 8, 32);
    rinv[r] = __builtin_amdgcn_rcpf(sum);
  }

#pragma unroll
  for (int nt = 0; nt < 4; ++nt) {                    // lm += col sums
    float cs = 0.f;
#pragma unroll
    for (int r = 0; r < 8; ++r) {
      int row = wave * 16 + r + 8 * g;
      if (row < 49) cs += S[nt][r] * rinv[r];
    }
    cs += __shfl_xor(cs, 16, 32);
    int key = nt * 16 + cl;
    if (lane < 16 && key < 49)
      atomicAdd(&lm[((size_t)b * 64 + win) * 49 + key], cs);
  }

#pragma unroll
  for (int nt = 0; nt < 4; ++nt)
#pragma unroll
    for (int r = 0; r < 8; ++r)
      Ps[wave][r + 8 * g][nt * 16 + cl] = (f16)S[nt][r];
  __syncthreads();

  v8f o[2] = {V8FZ, V8FZ};
#pragma unroll
  for (int kt = 0; kt < 2; ++kt) {
    const f16* pp = &Ps[wave][cl][kt * 32 + g * 16];
    v16h pa = cat8(*(const v8h*)pp, *(const v8h*)(pp + 8));
#pragma unroll
    for (int dt = 0; dt < 2; ++dt) {
      const f16* vp = Vt2 +
          ((((size_t)(b * 4 + h) * 64 + win) * 32 + dt * 16 + cl) * 64) +
          kt * 32 + g * 16;
      o[dt] = wmma32(pa, *(const v16h*)vp, o[dt]);
    }
  }
#pragma unroll
  for (int dt = 0; dt < 2; ++dt) {
#pragma unroll
    for (int r = 0; r < 8; ++r) {
      int row = wave * 16 + r + 8 * g;
      if (row < 49) {
        int ry = row / 7, rx = row % 7;
        int n = (wy * 7 + ry) * 56 + wx * 7 + rx;
        y[((size_t)b * 3136 + n) * 256 + 128 + h * 32 + dt * 16 + cl] =
            o[dt][r] * rinv[r];
      }
    }
  }
}

// mask = lm_pix/(4*49) + gm_pix/(4*3136); m1 row-major, m2 transposed
__global__ __launch_bounds__(256) void mask_kernel(
    const float* __restrict__ gm, const float* __restrict__ lm,
    float* __restrict__ m1, float* __restrict__ m2) {
  int b = blockIdx.x;
  for (int idx = threadIdx.x; idx < 3136; idx += 256) {
    int h = idx / 56, w = idx % 56;
    float gv = gm[b * 196 + (h >> 2) * 14 + (w >> 2)] * (1.f / (4.f * 3136.f));
    int wy = h / 7, qy = h % 7, wx = w / 7, qx = w % 7;
    float lv = lm[((size_t)b * 64 + wy * 8 + wx) * 49 + qy * 7 + qx] *
               (1.f / (4.f * 49.f));
    float mv = gv + lv;
    m1[(size_t)b * 3136 + h * 56 + w] = mv;
    m2[(size_t)b * 3136 + w * 56 + h] = mv;
  }
}

// ---------------------------------------------------------------------------
extern "C" void kernel_launch(void* const* d_in, const int* in_sizes, int n_in,
                              void* d_out, int out_size, void* d_ws, size_t ws_size,
                              hipStream_t stream) {
  (void)in_sizes; (void)n_in; (void)out_size; (void)ws_size;
  const float* x           = (const float*)d_in[0];
  const float* lepe_lin_w  = (const float*)d_in[3];
  const float* lepe_lin_b  = (const float*)d_in[4];
  const float* lepe_conv_w = (const float*)d_in[5];
  const float* lepe_conv_b = (const float*)d_in[6];
  const float* sr_w        = (const float*)d_in[7];
  const float* sr_b        = (const float*)d_in[8];
  const float* ln_g        = (const float*)d_in[9];
  const float* ln_b        = (const float*)d_in[10];
  const float* q1_w        = (const float*)d_in[11];
  const float* kv1_w       = (const float*)d_in[12];
  const float* q2_w        = (const float*)d_in[13];
  const float* kv2_w       = (const float*)d_in[14];
  const float* proj_w      = (const float*)d_in[15];
  const float* proj_b      = (const float*)d_in[16];
  float* out = (float*)d_out;

  // ---- workspace carve-up (256B aligned) ----
  char* wsb = (char*)d_ws;
  size_t off = 0;
  auto carve = [&](size_t bytes) -> void* {
    void* p = wsb + off;
    off += (bytes + 255) & ~(size_t)255;
    return p;
  };
  float* t_y   = (float*)carve((size_t)50176 * 256 * 4);  // t, later y
  float* lepe  = (float*)carve((size_t)50176 * 256 * 4);
  f16*   x16   = (f16*)  carve((size_t)50176 * 256 * 2);
  f16*   srA   = (f16*)  carve((size_t)3136 * 4096 * 2);  // later reused: yl16
  f16*   q1h   = (f16*)  carve((size_t)50176 * 128 * 2);
  f16*   q2h   = (f16*)  carve((size_t)50176 * 128 * 2);
  f16*   kv2h  = (f16*)  carve((size_t)50176 * 256 * 2);
  float* xs    = (float*)carve((size_t)3136 * 256 * 4);
  f16*   xsa16 = (f16*)  carve((size_t)3136 * 256 * 2);
  f16*   kv1h  = (f16*)  carve((size_t)3136 * 256 * 2);
  f16*   Vt1   = (f16*)  carve((size_t)16 * 4 * 32 * 224 * 2);
  f16*   Vt2   = (f16*)  carve((size_t)16 * 4 * 64 * 32 * 64 * 2);
  float* gm    = (float*)carve((size_t)16 * 196 * 4);
  float* lm    = (float*)carve((size_t)16 * 64 * 49 * 4);
  f16*   BpLep = (f16*)  carve((size_t)8 * 16 * 512 * 2);
  f16*   BpQ1  = (f16*)  carve((size_t)8 * 8 * 512 * 2);
  f16*   BpQ2  = (f16*)  carve((size_t)8 * 8 * 512 * 2);
  f16*   BpKV2 = (f16*)  carve((size_t)8 * 16 * 512 * 2);
  f16*   BpKV1 = (f16*)  carve((size_t)8 * 16 * 512 * 2);
  f16*   BpPrj = (f16*)  carve((size_t)8 * 16 * 512 * 2);
  f16*   BpSr  = (f16*)  carve((size_t)128 * 16 * 512 * 2);
  f16*   yl16  = srA;     // srA dead after sr GEMM; reuse for (y+lepe) f16

  hipMemsetAsync(gm, 0, (16 * 196 + 16 * 64 * 49) * sizeof(float), stream);

  // prep: f16 activations + fragment-packed weights
  cvt8<<<6272, 256, 0, stream>>>(x, x16, 50176 * 256 / 8);
  gather_srA<<<6272, 256, 0, stream>>>(x, srA);
  pack_wB<<<32, 256, 0, stream>>>(lepe_lin_w, BpLep, 8, 16, 256);
  pack_wB<<<16, 256, 0, stream>>>(q1_w, BpQ1, 8, 8, 128);
  pack_wB<<<16, 256, 0, stream>>>(q2_w, BpQ2, 8, 8, 128);
  pack_wB<<<32, 256, 0, stream>>>(kv2_w, BpKV2, 8, 16, 256);
  pack_wB<<<32, 256, 0, stream>>>(kv1_w, BpKV1, 8, 16, 256);
  pack_wB<<<32, 256, 0, stream>>>(proj_w, BpPrj, 8, 16, 256);
  pack_srwB<<<512, 256, 0, stream>>>(sr_w, BpSr);

  // lepe: linear (f32 out) -> depthwise conv
  gemm_rm<<<dim3(4, 784), 128, 0, stream>>>(x16, BpLep, lepe_lin_b, t_y,
                                            nullptr, 50176, 256, 256);
  lepe_conv<<<50176, 256, 0, stream>>>(t_y, lepe_conv_w, lepe_conv_b, lepe);

  // projections from x (f16 outputs feed attention directly)
  gemm_rm<<<dim3(2, 784), 128, 0, stream>>>(x16, BpQ1, nullptr, nullptr, q1h,
                                            50176, 128, 256);
  gemm_rm<<<dim3(2, 784), 128, 0, stream>>>(x16, BpQ2, nullptr, nullptr, q2h,
                                            50176, 128, 256);
  gemm_rm<<<dim3(4, 784), 128, 0, stream>>>(x16, BpKV2, nullptr, nullptr, kv2h,
                                            50176, 256, 256);

  // SR branch: im2col GEMM (K=4096) -> LN+GELU -> kv1
  gemm_rm<<<dim3(4, 49), 128, 0, stream>>>(srA, BpSr, sr_b, xs, nullptr,
                                           3136, 256, 4096);
  ln_gelu<<<3136, 256, 0, stream>>>(xs, ln_g, ln_b, xsa16);
  gemm_rm<<<dim3(4, 49), 128, 0, stream>>>(xsa16, BpKV1, nullptr, nullptr,
                                           kv1h, 3136, 256, 256);

  // transposed V buffers for P.V fragments
  pack_Vt1<<<224, 256, 0, stream>>>(kv1h, Vt1);
  pack_Vt2<<<4096, 256, 0, stream>>>(kv2h, Vt2);

  // attention: y(=t_y) cols 0..127 global, 128..255 local
  attn_global<<<dim3(49, 4, 16), 128, 0, stream>>>(q1h, kv1h, Vt1, t_y, gm);
  attn_local<<<dim3(64, 4, 16), 128, 0, stream>>>(q2h, kv2h, Vt2, t_y, lm);

  // out = (y + lepe) @ proj_w + proj_b
  addcvt8<<<6272, 256, 0, stream>>>(t_y, lepe, yl16, 50176 * 256 / 8);
  gemm_rm<<<dim3(4, 784), 128, 0, stream>>>(yl16, BpPrj, proj_b, out, nullptr,
                                            50176, 256, 256);

  // masks after the 50176x256 output block
  float* m1 = out + (size_t)50176 * 256;
  float* m2 = m1 + 50176;
  mask_kernel<<<16, 256, 0, stream>>>(gm, lm, m1, m2);
}
